// AdaptiveCombiner_72825465471276
// MI455X (gfx1250) — matrix operand order repaired and checked by reading, not compiled, not run on hardware
//
#include <hip/hip_runtime.h>
#include <hip/hip_bf16.h>

typedef _Float16 f16;
typedef __attribute__((ext_vector_type(16))) _Float16 v16h;
typedef __attribute__((ext_vector_type(8)))  float    v8f;
typedef __attribute__((ext_vector_type(4)))  float    v4f;

#define KNN   32
#define VOC   32000
#define NTOK  2048          // B*S
#define TPW   16            // tokens per wave (WMMA M dim)

// ---------------------------------------------------------------------------
// Kernel A: per-token feature build + MLP via WMMA + knn mixture weights.
// One wave (32 lanes) handles 16 tokens. Outputs w[token][32] and lam[token].
// ---------------------------------------------------------------------------
__global__ __launch_bounds__(32) void mlp_knn_kernel(
    const float* __restrict__ dist,  const int* __restrict__ vals,
    const float* __restrict__ W1,    const float* __restrict__ B1,
    const float* __restrict__ W2,    const float* __restrict__ B2,
    float* __restrict__ w_out,       float* __restrict__ lam_out)
{
    __shared__ float netin_s[TPW][64];   // [token][feature]
    __shared__ float h_s[TPW][32];       // hidden activations
    __shared__ float d2_s[TPW][16];      // layer-2 logits (cols 0..6 valid)
    __shared__ int   vals_s[32];
    __shared__ int   fo_s[32];

    const int lane = threadIdx.x;            // 0..31
    const int tok0 = blockIdx.x * TPW;

    // ---- net_in = [distances, distinct-nonzero prefix counts] ----
    for (int t = 0; t < TPW; ++t) {
        const int   tok = tok0 + t;
        const int   v   = vals[tok * KNN + lane];
        const float d   = dist[tok * KNN + lane];
        vals_s[lane] = v;
        __syncthreads();
        int dup = 0;
        for (int j = 0; j < 32; ++j) dup |= (j < lane) & (vals_s[j] == v);
        fo_s[lane] = (v != 0) && !dup;       // first occurrence of nonzero label
        __syncthreads();
        int c = 0;
        for (int j = 0; j <= lane; ++j) c += fo_s[j];  // inclusive prefix sum
        netin_s[t][lane]      = d;
        netin_s[t][32 + lane] = (float)c;
        __syncthreads();
    }

    const int lh = lane >> 4;    // lane half (0/1)
    const int m  = lane & 15;    // M row for A-frags, N col for B/C/D frags

    // ---- layer 1: (16x64) @ (64x32), f16 WMMA with f32 accumulate ----
    v8f acc0 = {}, acc1 = {};
    for (int ks = 0; ks < 2; ++ks) {                 // two K=32 chunks
        const int kbase = ks * 32;
        v16h a;
        for (int p = 0; p < 8; ++p) {                // A 16-bit 16x32 layout
            const int kp = kbase + ((p < 4) ? 2 * p : 16 + 2 * (p - 4)) + 8 * lh;
            a[2 * p]     = (f16)netin_s[m][kp];
            a[2 * p + 1] = (f16)netin_s[m][kp + 1];
        }
        v16h b0, b1f;
        for (int j = 0; j < 8; ++j) {                // B 16-bit 32x16 layout
            const int kk = kbase + ((j < 4) ? 2 * j : 16 + 2 * (j - 4)) + 8 * lh;
            b0[2 * j]      = (f16)W1[kk * 32 + m];
            b0[2 * j + 1]  = (f16)W1[(kk + 1) * 32 + m];
            b1f[2 * j]     = (f16)W1[kk * 32 + 16 + m];
            b1f[2 * j + 1] = (f16)W1[(kk + 1) * 32 + 16 + m];
        }
        acc0 = __builtin_amdgcn_wmma_f32_16x16x32_f16(false, a, false, b0,  (short)0, acc0, false, false);
        acc1 = __builtin_amdgcn_wmma_f32_16x16x32_f16(false, a, false, b1f, (short)0, acc1, false, false);
    }
    for (int j = 0; j < 8; ++j) {                    // D: VGPR j -> M=j+8*lh, N=m
        const int mm = j + 8 * lh;
        h_s[mm][m]      = tanhf(acc0[j] + B1[m]);
        h_s[mm][16 + m] = tanhf(acc1[j] + B1[16 + m]);
    }
    __syncthreads();

    // ---- layer 2: (16x32) @ (32x7 zero-padded to 16), 1 WMMA ----
    v16h a2;
    for (int p = 0; p < 8; ++p) {
        const int kp = ((p < 4) ? 2 * p : 16 + 2 * (p - 4)) + 8 * lh;
        a2[2 * p]     = (f16)h_s[m][kp];
        a2[2 * p + 1] = (f16)h_s[m][kp + 1];
    }
    v16h b2f;
    for (int j = 0; j < 8; ++j) {
        const int kk = ((j < 4) ? 2 * j : 16 + 2 * (j - 4)) + 8 * lh;
        b2f[2 * j]     = (m < 7) ? (f16)W2[kk * 7 + m]       : (f16)0.0f;
        b2f[2 * j + 1] = (m < 7) ? (f16)W2[(kk + 1) * 7 + m] : (f16)0.0f;
    }
    v8f acc2 = {};
    acc2 = __builtin_amdgcn_wmma_f32_16x16x32_f16(false, a2, false, b2f, (short)0, acc2, false, false);
    for (int j = 0; j < 8; ++j) {
        const int mm = j + 8 * lh;
        d2_s[mm][m] = acc2[j] + ((m < 7) ? B2[m] : 0.0f);
    }
    __syncthreads();

    // ---- per-token epilogue: softmax-7, lambda, knn mixture weights ----
    if (lane < TPW) {
        const int tok = tok0 + lane;
        float lg[7];
        float mx = -3.4e38f;
        for (int r = 0; r < 7; ++r) { lg[r] = d2_s[lane][r]; mx = fmaxf(mx, lg[r]); }
        float s = 0.f;
        for (int r = 0; r < 7; ++r) { lg[r] = __expf(lg[r] - mx); s += lg[r]; }
        const float inv = 1.f / s;
        const float lam = 1.f - lg[0] * inv;
        float ksel[6];
        for (int r = 0; r < 6; ++r) ksel[r] = lg[r + 1] * inv;

        // knn_weight[r][k] = softmax_k( -(d[k]*mask[r][k]) / 10 )
        // distances are sorted ascending -> row max logit is exactly -d0/10.
        const float d0 = netin_s[lane][0];
        float rs[6];
        for (int r = 0; r < 6; ++r) {
            const int lim = 1 << r;
            float acc = 0.f;
            for (int k = 0; k < KNN; ++k) {
                const float dk = netin_s[lane][k];
                const float mk = (k < lim) ? 1.0f : 1000.0f;
                acc += __expf((d0 - dk * mk) * 0.1f);
            }
            rs[r] = 1.f / acc;
        }
        for (int k = 0; k < KNN; ++k) {
            const float dk = netin_s[lane][k];
            float w = 0.f;
            for (int r = 0; r < 6; ++r) {
                const float mk = (k < (1 << r)) ? 1.0f : 1000.0f;
                w += ksel[r] * rs[r] * __expf((d0 - dk * mk) * 0.1f);
            }
            w_out[tok * KNN + k] = w;
        }
        lam_out[tok] = lam;
    }
}

// ---------------------------------------------------------------------------
// Kernel B: per-token row softmax over V=32000, blend, scatter knn weights.
// One 256-thread block per token; row re-reads hit L2; output streamed NT.
// ---------------------------------------------------------------------------
__global__ __launch_bounds__(256) void softmax_combine_kernel(
    const float* __restrict__ logit, const int* __restrict__ vals,
    const float* __restrict__ w_ws,  const float* __restrict__ lam_ws,
    float* __restrict__ out)
{
    const int    t    = blockIdx.x;
    const int    tid  = threadIdx.x;
    const size_t base = (size_t)t * VOC;
    const v4f* __restrict__ row4 = (const v4f*)(logit + base);
    v4f*       __restrict__ out4 = (v4f*)(out + base);
    const int N4 = VOC / 4;          // 8000 float4s per row

    __shared__ float red[8];
    __shared__ float bcast;

    // ---- pass 1: row max ----
    float mx = -3.4e38f;
    for (int i = tid; i < N4; i += 256) {
        v4f x = row4[i];
        mx = fmaxf(fmaxf(mx, x.x), fmaxf(x.y, fmaxf(x.z, x.w)));
    }
    for (int off = 16; off; off >>= 1) mx = fmaxf(mx, __shfl_xor(mx, off, 32));
    if ((tid & 31) == 0) red[tid >> 5] = mx;
    __syncthreads();
    if (tid == 0) {
        float v = red[0];
        for (int i = 1; i < 8; ++i) v = fmaxf(v, red[i]);
        bcast = v;
    }
    __syncthreads();
    mx = bcast;

    // ---- pass 2: sum of exp (re-read hits L2) ----
    float sum = 0.f;
    for (int i = tid; i < N4; i += 256) {
        v4f x = row4[i];
        sum += __expf(x.x - mx) + __expf(x.y - mx) + __expf(x.z - mx) + __expf(x.w - mx);
    }
    for (int off = 16; off; off >>= 1) sum += __shfl_xor(sum, off, 32);
    if ((tid & 31) == 0) red[tid >> 5] = sum;
    __syncthreads();
    if (tid == 0) {
        float v = 0.f;
        for (int i = 1; i < 8; ++i) v += red[i];
        bcast = v + red[0];
    }
    __syncthreads();
    sum = bcast;

    const float lam   = lam_ws[t];
    const float scale = (1.0f - lam) / sum;

    // ---- pass 3: neural part, streamed with non-temporal b128 stores ----
    for (int i = tid; i < N4; i += 256) {
        v4f x = row4[i];
        v4f o;
        o.x = __expf(x.x - mx) * scale;
        o.y = __expf(x.y - mx) * scale;
        o.z = __expf(x.z - mx) * scale;
        o.w = __expf(x.w - mx) * scale;
        __builtin_nontemporal_store(o, &out4[i]);
    }

    __threadfence();      // streamed stores visible before RMW scatter
    __syncthreads();

    // ---- pass 4: scatter knn mass (duplicate labels accumulate via atomics) ----
    if (tid < KNN) {
        const int   v   = vals[t * KNN + tid];
        const float add = lam * w_ws[t * KNN + tid];
        atomicAdd(out + base + v, add);
    }
}

// ---------------------------------------------------------------------------
extern "C" void kernel_launch(void* const* d_in, const int* in_sizes, int n_in,
                              void* d_out, int out_size, void* d_ws, size_t ws_size,
                              hipStream_t stream)
{
    const float* dist  = (const float*)d_in[0];   // (B,S,K)
    const int*   vals  = (const int*)  d_in[1];   // (B,S,K,1) int32
    const float* logit = (const float*)d_in[2];   // (B,S,V)
    const float* W1    = (const float*)d_in[3];   // (64,32)
    const float* B1    = (const float*)d_in[4];   // (32)
    const float* W2    = (const float*)d_in[5];   // (32,7)
    const float* B2    = (const float*)d_in[6];   // (7)
    float*       out   = (float*)d_out;

    const int T = in_sizes[0] / KNN;              // 2048 tokens

    float* w_ws   = (float*)d_ws;                 // T*32 floats
    float* lam_ws = w_ws + (size_t)T * KNN;       // T floats

    mlp_knn_kernel<<<T / TPW, 32, 0, stream>>>(dist, vals, W1, B1, W2, B2, w_ws, lam_ws);
    softmax_combine_kernel<<<T, 256, 0, stream>>>(logit, vals, w_ws, lam_ws, out);
}
